// DeepSeekMoE_4879082848971
// MI455X (gfx1250) — matrix operand order repaired and checked by reading, not compiled, not run on hardware
//
#include <hip/hip_runtime.h>
#include <math.h>

// ---------------------------------------------------------------------------
// MI455X (gfx1250) DeepSeek-MoE. wave32, WMMA bf16 16x16x32, f32 accum.
// Compute-bound (~600 FLOP/byte) -> bf16 matrix pipes. Double-buffered LDS,
// 1 barrier / K-step, async global->LDS for bf16 activation tiles.
// All 8 experts run concurrently (gridDim.z=8): every write lands at a
// globally unique slot row (2*token + topk_slot), so no atomics are needed
// and the result is bitwise deterministic.
// ---------------------------------------------------------------------------

typedef __attribute__((ext_vector_type(16))) __bf16 v16bf;
typedef __attribute__((ext_vector_type(8)))  __bf16 v8bf;
typedef __attribute__((ext_vector_type(8)))  float  v8f;
typedef __attribute__((ext_vector_type(4)))  float  v4f;
typedef __attribute__((ext_vector_type(4)))  int    v4i;

#define D_MODEL 1024
#define D_FF    2048
#define N_EXP   8
#define T_TOK   4096
#define TM 64
#define TN 64
#define TK 32
#define LDSROW 40   // 32 + 8 pad bf16 -> 80B row stride (16B aligned, bank-spread)

#if defined(__AMDGCN__) && __has_builtin(__builtin_amdgcn_global_load_async_to_lds_b128)
#define USE_ASYNC_LDS 1
#endif

#ifdef USE_ASYNC_LDS
__device__ __forceinline__ void async_cp16(const void* g, void* l) {
    // GLOBAL_LOAD_ASYNC_TO_LDS_B128: 16B global -> LDS, tracked by ASYNCcnt
    __builtin_amdgcn_global_load_async_to_lds_b128((v4i*)g, (v4i*)l, 0, 0);
}
__device__ __forceinline__ void wait_async0() {
#if __has_builtin(__builtin_amdgcn_s_wait_asynccnt)
    __builtin_amdgcn_s_wait_asynccnt(0);
#else
    asm volatile("s_wait_asynccnt 0x0" ::: "memory");
#endif
}
#endif

// ---------------------------------------------------------------------------
// Kernel 0: zero per-expert counters
// ---------------------------------------------------------------------------
__global__ void moe_init(int* counts) {
    if (threadIdx.x < N_EXP) counts[threadIdx.x] = 0;
}

// ---------------------------------------------------------------------------
// Kernel 1: router. logits -> softmax -> top2 -> renormalized softmax.
// List entry = token | (k << 16); per-token combine weights in wtk[2t+k].
// Slot row 2t+k is globally unique -> downstream writes are race-free.
// ---------------------------------------------------------------------------
__global__ __launch_bounds__(256)
void moe_router(const float* __restrict__ X, const float* __restrict__ Rw,
                const float* __restrict__ bias,
                int* __restrict__ counts, int* __restrict__ lists,
                float* __restrict__ wtk) {
    int t = blockIdx.x * blockDim.x + threadIdx.x;
    if (t >= T_TOK) return;
    const float* x = X + (size_t)t * D_MODEL;

    float acc[N_EXP];
#pragma unroll
    for (int e = 0; e < N_EXP; ++e) acc[e] = 0.f;
    for (int k = 0; k < D_MODEL; k += 4) {
        v4f xv = *(const v4f*)(x + k);
#pragma unroll
        for (int e = 0; e < N_EXP; ++e) {
            v4f wv = *(const v4f*)(Rw + (size_t)e * D_MODEL + k);
            acc[e] += xv.x * wv.x + xv.y * wv.y + xv.z * wv.z + xv.w * wv.w;
        }
    }
    float l[N_EXP], mx = -1e30f;
#pragma unroll
    for (int e = 0; e < N_EXP; ++e) { l[e] = acc[e] + bias[e]; mx = fmaxf(mx, l[e]); }
    float p[N_EXP], s = 0.f;
#pragma unroll
    for (int e = 0; e < N_EXP; ++e) { p[e] = __expf(l[e] - mx); s += p[e]; }
    float inv_s = 1.f / s;
#pragma unroll
    for (int e = 0; e < N_EXP; ++e) p[e] *= inv_s;

    int i0 = 0;
#pragma unroll
    for (int e = 1; e < N_EXP; ++e) if (p[e] > p[i0]) i0 = e;
    int i1 = (i0 == 0) ? 1 : 0;
#pragma unroll
    for (int e = 0; e < N_EXP; ++e) if (e != i0 && p[e] > p[i1]) i1 = e;

    float e1 = __expf(p[i1] - p[i0]);          // softmax over {p[i0], p[i1]}
    float w0 = 1.f / (1.f + e1);
    float w1 = e1 * w0;

    wtk[2 * t + 0] = w0;
    wtk[2 * t + 1] = w1;
    int s0 = atomicAdd(&counts[i0], 1);
    lists[i0 * T_TOK + s0] = t;                // k = 0
    int s1 = atomicAdd(&counts[i1], 1);
    lists[i1 * T_TOK + s1] = t | (1 << 16);    // k = 1
}

// ---------------------------------------------------------------------------
// WMMA helpers
// ---------------------------------------------------------------------------
__device__ __forceinline__ v8f wmma_bf16(v16bf a, v16bf b, v8f c) {
    return __builtin_amdgcn_wmma_f32_16x16x32_bf16(
        false, a, false, b, (short)0, c, false, false);
}
__device__ __forceinline__ v16bf cat16(v8bf lo, v8bf hi) {
    return __builtin_shufflevector(lo, hi, 0,1,2,3,4,5,6,7,8,9,10,11,12,13,14,15);
}
__device__ __forceinline__ v8bf cvt8(const float* r) {
    v8bf o;
#pragma unroll
    for (int i = 0; i < 8; ++i) o[i] = (__bf16)r[i];
    return o;
}
__device__ __forceinline__ float fast_rcp(float x) {
#if defined(__AMDGCN__)
    return __builtin_amdgcn_rcpf(x);     // v_rcp_f32
#else
    return 1.f / x;
#endif
}
__device__ __forceinline__ int slot_of(int entry) {       // entry = t | k<<16
    return 2 * (entry & 0xFFFF) + (entry >> 16);
}

// ---------------------------------------------------------------------------
// Kernel 2: fused gate/up GEMM + SwiGLU. G[row] = silu(X Wg^T) * (X Wu^T).
// Shared mode (lists==null): rows = tokens, G row = token.
// Expert mode: e = blockIdx.z, rows gathered via lists, G row = 2*tok + k
// (globally unique -> all experts run concurrently).
// ---------------------------------------------------------------------------
__global__ __launch_bounds__(256)
void moe_gateup(const float* __restrict__ X, const float* __restrict__ Wg,
                const float* __restrict__ Wu, __bf16* __restrict__ G,
                const int* __restrict__ lists, const int* __restrict__ counts) {
    __shared__ __bf16 As[2][TM][LDSROW];
    __shared__ __bf16 Bg[2][TN][LDSROW];
    __shared__ __bf16 Bu[2][TN][LDSROW];

    const int mblk = blockIdx.y, nblk = blockIdx.x, e = blockIdx.z;
    const int* list = nullptr;
    int cnt = T_TOK;
    if (lists) {
        list = lists + e * T_TOK;
        cnt = counts[e];
        const size_t wo = (size_t)e * D_FF * D_MODEL;
        Wg += wo; Wu += wo;
    }
    if (mblk * TM >= cnt) return;   // uniform early exit (expert tail blocks)

    const int tid  = threadIdx.x;
    const int lrow = tid >> 2;        // 0..63 staged row
    const int col0 = (tid & 3) * 8;   // 0,8,16,24 within K-step

    int growA = mblk * TM + lrow;
    int tok = growA;
    if (list) {
        int en = (growA < cnt) ? list[growA] : list[0];   // clamp padded rows
        tok = en & 0xFFFF;
    }
    const float* arow = X  + (size_t)tok * D_MODEL;
    const float* grow = Wg + (size_t)(nblk * TN + lrow) * D_MODEL;
    const float* urow = Wu + (size_t)(nblk * TN + lrow) * D_MODEL;

    const int wave = tid >> 5, lane = tid & 31;
    const int h = lane >> 4, ln = lane & 15;
    const int wm = wave & 1;    // M half (32 rows)
    const int wn = wave >> 1;   // N quarter (16 cols)

    v8f accg[2] = { {}, {} }, accu[2] = { {}, {} };

    float ra[8], rg[8], ru[8];
#pragma unroll
    for (int i = 0; i < 8; ++i) {           // k0 = 0
        ra[i] = arow[col0 + i];
        rg[i] = grow[col0 + i];
        ru[i] = urow[col0 + i];
    }
    *(v8bf*)&As[0][lrow][col0] = cvt8(ra);
    *(v8bf*)&Bg[0][lrow][col0] = cvt8(rg);
    *(v8bf*)&Bu[0][lrow][col0] = cvt8(ru);
    __syncthreads();

    int cur = 0;
    for (int k0 = 0; k0 < D_MODEL; k0 += TK) {
        const bool more = (k0 + TK) < D_MODEL;
        if (more) {
#pragma unroll
            for (int i = 0; i < 8; ++i) {   // register prefetch of next K tile
                ra[i] = arow[k0 + TK + col0 + i];
                rg[i] = grow[k0 + TK + col0 + i];
                ru[i] = urow[k0 + TK + col0 + i];
            }
        }
        // A frag: lane h*16+m -> K [8h,8h+8) ++ [16+8h,24+8h)
        // B frag: lane h*16+n -> K [8h,8h+8) ++ [16+8h,24+8h)
        const int r0 = wm * 32 + ln, r1 = r0 + 16, bn = wn * 16 + ln;
        v16bf a0 = cat16(*(v8bf*)&As[cur][r0][8*h], *(v8bf*)&As[cur][r0][16 + 8*h]);
        v16bf a1 = cat16(*(v8bf*)&As[cur][r1][8*h], *(v8bf*)&As[cur][r1][16 + 8*h]);
        v16bf bg = cat16(*(v8bf*)&Bg[cur][bn][8*h], *(v8bf*)&Bg[cur][bn][16 + 8*h]);
        v16bf bu = cat16(*(v8bf*)&Bu[cur][bn][8*h], *(v8bf*)&Bu[cur][bn][16 + 8*h]);
        accg[0] = wmma_bf16(a0, bg, accg[0]);
        accg[1] = wmma_bf16(a1, bg, accg[1]);
        accu[0] = wmma_bf16(a0, bu, accu[0]);
        accu[1] = wmma_bf16(a1, bu, accu[1]);
        if (more) {                          // stage into the other buffer
            *(v8bf*)&As[cur ^ 1][lrow][col0] = cvt8(ra);
            *(v8bf*)&Bg[cur ^ 1][lrow][col0] = cvt8(rg);
            *(v8bf*)&Bu[cur ^ 1][lrow][col0] = cvt8(ru);
        }
        __syncthreads();
        cur ^= 1;
    }

    // epilogue: C layout lane h*16+n, VGPR r -> (M = r + 8h, N = n)
    const int ncol = nblk * TN + wn * 16 + ln;
#pragma unroll
    for (int s = 0; s < 2; ++s) {
#pragma unroll
        for (int r = 0; r < 8; ++r) {
            int row = mblk * TM + wm * 32 + s * 16 + r + 8 * h;
            float hg = accg[s][r], hu = accu[s][r];
            float g = hg * fast_rcp(1.f + __expf(-hg)) * hu;   // silu(hg)*hu
            if (list) {
                if (row < cnt) G[(size_t)slot_of(list[row]) * D_FF + ncol] = (__bf16)g;
            } else {
                G[(size_t)row * D_FF + ncol] = (__bf16)g;
            }
        }
    }
}

// ---------------------------------------------------------------------------
// Kernel 3: down projection. Y = G Wd^T.
// Shared mode: A row = token row of G, dst[row] = y (initializes d_out).
// Expert mode (e = blockIdx.z): A rows gathered by slot, dst = Y buffer,
// Y[slot] = y (unique slot -> plain store, concurrent experts).
// A tile (bf16) via async DMA when available; B (fp32) converted in regs.
// ---------------------------------------------------------------------------
__global__ __launch_bounds__(256)
void moe_down(const __bf16* __restrict__ G, const float* __restrict__ Wd,
              float* __restrict__ dst,
              const int* __restrict__ lists, const int* __restrict__ counts) {
    __shared__ __bf16 As[2][TM][LDSROW];
    __shared__ __bf16 Bs[2][TN][LDSROW];

    const int mblk = blockIdx.y, nblk = blockIdx.x, e = blockIdx.z;
    const int* list = nullptr;
    int cnt = T_TOK;
    if (lists) {
        list = lists + e * T_TOK;
        cnt = counts[e];
        Wd += (size_t)e * D_MODEL * D_FF;
    }
    if (mblk * TM >= cnt) return;

    const int tid  = threadIdx.x;
    const int lrow = tid >> 2;
    const int col0 = (tid & 3) * 8;

    int growA = mblk * TM + lrow;
    int arowid = growA;
    if (list) arowid = slot_of((growA < cnt) ? list[growA] : list[0]);
    const __bf16* arow = G  + (size_t)arowid * D_FF;               // bf16 rows
    const float*  brow = Wd + (size_t)(nblk * TN + lrow) * D_FF;

    const int wave = tid >> 5, lane = tid & 31;
    const int h = lane >> 4, ln = lane & 15;
    const int wm = wave & 1, wn = wave >> 1;

    v8f acc[2] = { {}, {} };

    // ---- stage k0 = 0 into buffer 0 ----
#ifdef USE_ASYNC_LDS
    async_cp16(arow + col0, &As[0][lrow][col0]);
#else
    v8bf rav = *(const v8bf*)(arow + col0);
    *(v8bf*)&As[0][lrow][col0] = rav;
#endif
    float rb[8];
#pragma unroll
    for (int i = 0; i < 8; ++i) rb[i] = brow[col0 + i];
    *(v8bf*)&Bs[0][lrow][col0] = cvt8(rb);
#ifdef USE_ASYNC_LDS
    wait_async0();
#endif
    __syncthreads();

    int cur = 0;
    for (int k0 = 0; k0 < D_FF; k0 += TK) {
        const bool more = (k0 + TK) < D_FF;
        if (more) {
            // issue next A tile early: DMA overlaps the WMMAs below
#ifdef USE_ASYNC_LDS
            async_cp16(arow + k0 + TK + col0, &As[cur ^ 1][lrow][col0]);
#else
            rav = *(const v8bf*)(arow + k0 + TK + col0);
#endif
#pragma unroll
            for (int i = 0; i < 8; ++i) rb[i] = brow[k0 + TK + col0 + i];
        }
        const int r0 = wm * 32 + ln, r1 = r0 + 16, bn = wn * 16 + ln;
        v16bf a0 = cat16(*(v8bf*)&As[cur][r0][8*h], *(v8bf*)&As[cur][r0][16 + 8*h]);
        v16bf a1 = cat16(*(v8bf*)&As[cur][r1][8*h], *(v8bf*)&As[cur][r1][16 + 8*h]);
        v16bf b  = cat16(*(v8bf*)&Bs[cur][bn][8*h], *(v8bf*)&Bs[cur][bn][16 + 8*h]);
        acc[0] = wmma_bf16(a0, b, acc[0]);
        acc[1] = wmma_bf16(a1, b, acc[1]);
        if (more) {
#ifndef USE_ASYNC_LDS
            *(v8bf*)&As[cur ^ 1][lrow][col0] = rav;
#endif
            *(v8bf*)&Bs[cur ^ 1][lrow][col0] = cvt8(rb);
#ifdef USE_ASYNC_LDS
            wait_async0();
#endif
        }
        __syncthreads();
        cur ^= 1;
    }

    const int ncol = nblk * TN + wn * 16 + ln;
#pragma unroll
    for (int s = 0; s < 2; ++s) {
#pragma unroll
        for (int r = 0; r < 8; ++r) {
            int row = mblk * TM + wm * 32 + s * 16 + r + 8 * h;
            float y = acc[s][r];
            if (list) {
                if (row < cnt)
                    dst[(size_t)slot_of(list[row]) * D_MODEL + ncol] = y;
            } else {
                dst[(size_t)row * D_MODEL + ncol] = y;
            }
        }
    }
}

// ---------------------------------------------------------------------------
// Kernel 4: combine. out[t] = shared_out[t] + w0*Y[2t] + w1*Y[2t+1]
// Fixed order -> bitwise deterministic.
// ---------------------------------------------------------------------------
__global__ __launch_bounds__(256)
void moe_combine(const float* __restrict__ Y, const float* __restrict__ wtk,
                 float* __restrict__ out) {
    int idx = blockIdx.x * blockDim.x + threadIdx.x;   // T*D/4 lanes of float4
    int t = idx >> (10 - 2);                           // D/4 = 256 chunks/token
    int c = (idx & 255) * 4;
    float w0 = wtk[2 * t], w1 = wtk[2 * t + 1];
    v4f y0 = *(const v4f*)(Y + (size_t)(2 * t)     * D_MODEL + c);
    v4f y1 = *(const v4f*)(Y + (size_t)(2 * t + 1) * D_MODEL + c);
    float* o = out + (size_t)t * D_MODEL + c;
    v4f ov = *(const v4f*)o;
    ov += w0 * y0 + w1 * y1;
    *(v4f*)o = ov;
}

// ---------------------------------------------------------------------------
// Launch: init -> router -> shared(gateup, down->out) ->
//         all-experts gateup (z=8) -> all-experts down->Y (z=8) -> combine
// ---------------------------------------------------------------------------
extern "C" void kernel_launch(void* const* d_in, const int* in_sizes, int n_in,
                              void* d_out, int out_size, void* d_ws, size_t ws_size,
                              hipStream_t stream) {
    const float* X       = (const float*)d_in[0];
    const float* Rw      = (const float*)d_in[1];
    const float* ebias   = (const float*)d_in[2];
    const float* sw_gate = (const float*)d_in[3];
    const float* sw_up   = (const float*)d_in[4];
    const float* sw_down = (const float*)d_in[5];
    const float* ew_gate = (const float*)d_in[6];
    const float* ew_up   = (const float*)d_in[7];
    const float* ew_down = (const float*)d_in[8];
    float* out = (float*)d_out;

    char* w = (char*)d_ws;
    int*    counts = (int*)w;                                   //    32 B
    int*    lists  = (int*)(w + 256);                           // 128 KiB
    float*  wtk    = (float*)(w + 256 + N_EXP * T_TOK * 4);     //  32 KiB
    float*  Y      = (float*)(w + 164096);                      //  32 MiB (8192 x 1024 f32)
    __bf16* G      = (__bf16*)(w + 164096 + (size_t)2 * T_TOK * D_MODEL * 4); // 32 MiB

    moe_init<<<1, 32, 0, stream>>>(counts);
    moe_router<<<T_TOK / 256, 256, 0, stream>>>(X, Rw, ebias, counts, lists, wtk);

    // shared expert: G rows = tokens; down initializes d_out
    moe_gateup<<<dim3(D_FF / TN, T_TOK / TM, 1), 256, 0, stream>>>(
        X, sw_gate, sw_up, G, nullptr, nullptr);
    moe_down<<<dim3(D_MODEL / TN, T_TOK / TM, 1), 256, 0, stream>>>(
        G, sw_down, out, nullptr, nullptr);

    // all 8 specialists concurrently: unique slot rows, no races
    moe_gateup<<<dim3(D_FF / TN, T_TOK / TM, N_EXP), 256, 0, stream>>>(
        X, ew_gate, ew_up, G, lists, counts);
    moe_down<<<dim3(D_MODEL / TN, T_TOK / TM, N_EXP), 256, 0, stream>>>(
        G, ew_down, Y, lists, counts);

    moe_combine<<<(T_TOK * D_MODEL / 4) / 256, 256, 0, stream>>>(Y, wtk, out);
}
